// BaseRGCN_72086731096972
// MI455X (gfx1250) — matrix operand order repaired and checked by reading, not compiled, not run on hardware
//
#include <hip/hip_runtime.h>

#define N_NODES   50000
#define N_EDGES   640000
#define HD        128
#define N_RELS    16
#define N_BASES   8

typedef __attribute__((ext_vector_type(16))) _Float16 v16h;
typedef __attribute__((ext_vector_type(8)))  _Float16 v8h;
typedef __attribute__((ext_vector_type(8)))  float    v8f;

// ---------------- init: zero accumulators + histogram counters ----------------
__global__ void init_zero_kernel(float* __restrict__ h1acc, float* __restrict__ out,
                                 int* __restrict__ counts, int nh) {
  int idx = blockIdx.x * blockDim.x + threadIdx.x;
  if (idx < nh)            h1acc[idx] = 0.0f;
  else if (idx < 2 * nh)   out[idx - nh] = 0.0f;
  if (idx < N_RELS)        counts[idx] = 0;
}

// ------------- W[r] = sum_b coef[r,b] * V[b]  (stored transposed, f16) --------
// Wt layout: [r][o][i]  (o = output col = WMMA "N", i = input = WMMA "K")
__global__ void build_w_kernel(const float* __restrict__ V,
                               const float* __restrict__ coef,
                               _Float16* __restrict__ Wt) {
  int idx = blockIdx.x * blockDim.x + threadIdx.x;
  if (idx >= N_RELS * HD * HD) return;
  int r = idx >> 14;
  int o = (idx >> 7) & 127;
  int i = idx & 127;
  float s = 0.0f;
#pragma unroll
  for (int b = 0; b < N_BASES; ++b)
    s += coef[r * N_BASES + b] * V[b * HD * HD + i * HD + o];
  Wt[idx] = (_Float16)s;
}

// ---------------- counting sort of edges by relation --------------------------
__global__ void hist_kernel(const int* __restrict__ etype, int* __restrict__ counts) {
  int e = blockIdx.x * blockDim.x + threadIdx.x;
  if (e < N_EDGES) atomicAdd(&counts[etype[e]], 1);
}

__global__ void prefix_kernel(const int* __restrict__ counts,
                              int* __restrict__ off, int* __restrict__ cursor) {
  if (threadIdx.x == 0 && blockIdx.x == 0) {
    int run = 0;
    for (int r = 0; r < N_RELS; ++r) { off[r] = run; cursor[r] = run; run += counts[r]; }
  }
}

__global__ void scatter_kernel(const int* __restrict__ etype,
                               int* __restrict__ cursor, int* __restrict__ order) {
  int e = blockIdx.x * blockDim.x + threadIdx.x;
  if (e < N_EDGES) {
    int p = atomicAdd(&cursor[etype[e]], 1);
    order[p] = e;
  }
}

// ---------------- gathered WMMA GEMM + norm-scaled atomic scatter -------------
// block = 256 threads (8 wave32); tile = 64 edges x 128 out; K = 128
__global__ __launch_bounds__(256)
void rgcn_gemm_kernel(const float* __restrict__ Asrc,      // [N,128] node features (f32)
                      const _Float16* __restrict__ Wt,     // [R][128 n][128 k] f16
                      const int* __restrict__ order,       // edges sorted by relation
                      const int* __restrict__ rel_off,
                      const int* __restrict__ rel_cnt,
                      const int* __restrict__ src,
                      const int* __restrict__ dst,
                      const float* __restrict__ norm,
                      float* __restrict__ dest) {          // [N,128] accumulator
  __shared__ alignas(16) _Float16 As[64][136];   // gathered rows, f16, padded
  __shared__ alignas(16) _Float16 Ws[128][136];  // Wt[r] tile: [n][k], padded
  __shared__ float snorm[64];
  __shared__ int   sdst[64];

  const int r   = blockIdx.y;
  const int cnt = rel_cnt[r];
  const int m0  = blockIdx.x * 64;
  if (m0 >= cnt) return;
  const int base = rel_off[r];
  const int t    = threadIdx.x;

  // stage W tile (2048 uint4 = 32 KB)
  {
    const uint4* wg = (const uint4*)(Wt + (size_t)r * HD * HD);
#pragma unroll
    for (int i = 0; i < 8; ++i) {
      int lin = t + i * 256;            // 0..2047
      int row = lin >> 4, c8 = lin & 15;
      *(uint4*)&Ws[row][c8 * 8] = wg[lin];
    }
  }

  // gather A rows (4 threads per row, 32 floats each), f32 -> f16
  {
    int row = t >> 2, q = t & 3;
    int m = m0 + row;
    int e = (m < cnt) ? order[base + m] : -1;
    if (q == 0) {
      snorm[row] = (e >= 0) ? norm[e] : 0.0f;
      sdst[row]  = (e >= 0) ? dst[e]  : 0;
    }
    if (e >= 0) {
      const float4* hp = (const float4*)(Asrc + (size_t)src[e] * HD + q * 32);
#pragma unroll
      for (int j = 0; j < 8; ++j) {
        float4 v = hp[j];
        int c = q * 32 + j * 4;
        As[row][c + 0] = (_Float16)v.x;
        As[row][c + 1] = (_Float16)v.y;
        As[row][c + 2] = (_Float16)v.z;
        As[row][c + 3] = (_Float16)v.w;
      }
    } else {
      int c = q * 32;
#pragma unroll
      for (int j = 0; j < 32; ++j) As[row][c + j] = (_Float16)0.0f;
    }
  }
  __syncthreads();

  const int lane = t & 31;
  const int wv   = t >> 5;        // wave id 0..7 -> N slab
  const int lr   = lane & 15;
  const int g    = lane >> 4;     // lane group
  const int ncol = wv * 16 + lr;  // this lane's output column

  v8f acc[4] = {v8f{}, v8f{}, v8f{}, v8f{}};

#pragma unroll
  for (int kt = 0; kt < 4; ++kt) {
    const int k0 = kt * 32;
    // B fragment: column ncol, K = k0 + {e + 8g (e<8) ; e+8 + 8g (e>=8)}
    v8h blo = *(const v8h*)&Ws[ncol][k0 + 8 * g];
    v8h bhi = *(const v8h*)&Ws[ncol][k0 + 16 + 8 * g];
    v16h bf = __builtin_shufflevector(blo, bhi, 0,1,2,3,4,5,6,7,8,9,10,11,12,13,14,15);
#pragma unroll
    for (int mt = 0; mt < 4; ++mt) {
      // A fragment: row mt*16+lr, K = k0 + {e + 8g (e<8) ; e+8 + 8g (e>=8)}
      const int ar = mt * 16 + lr;
      v8h alo = *(const v8h*)&As[ar][k0 + 8 * g];
      v8h ahi = *(const v8h*)&As[ar][k0 + 16 + 8 * g];
      v16h af = __builtin_shufflevector(alo, ahi, 0,1,2,3,4,5,6,7,8,9,10,11,12,13,14,15);
      acc[mt] = __builtin_amdgcn_wmma_f32_16x16x32_f16(
          false, af, false, bf, (short)0, acc[mt], false, false);
    }
  }

  // epilogue: msg = acc * norm, atomic scatter-add to dest[dst]
#pragma unroll
  for (int mt = 0; mt < 4; ++mt) {
#pragma unroll
    for (int j = 0; j < 8; ++j) {
      int m = mt * 16 + g * 8 + j;      // C/D layout: M = j + 8*laneGroup
      if (m0 + m < cnt) {
        float val = acc[mt][j] * snorm[m];
        atomicAdd(dest + (size_t)sdst[m] * HD + ncol, val);
      }
    }
  }
}

// ---------------- epilogues -----------------
__global__ void bias_relu_kernel(float* __restrict__ h1, const float* __restrict__ bias, int nh) {
  int idx = blockIdx.x * blockDim.x + threadIdx.x;
  if (idx < nh) {
    float v = h1[idx] + bias[idx & 127];
    h1[idx] = v > 0.0f ? v : 0.0f;
  }
}

__global__ void bias_kernel(float* __restrict__ out, const float* __restrict__ bias, int nh) {
  int idx = blockIdx.x * blockDim.x + threadIdx.x;
  if (idx < nh) out[idx] += bias[idx & 127];
}

// ---------------- host launch -----------------
extern "C" void kernel_launch(void* const* d_in, const int* in_sizes, int n_in,
                              void* d_out, int out_size, void* d_ws, size_t ws_size,
                              hipStream_t stream) {
  const float* h     = (const float*)d_in[0];
  const float* norm  = (const float*)d_in[1];
  const int*   src   = (const int*)d_in[2];
  const int*   dst   = (const int*)d_in[3];
  const int*   etype = (const int*)d_in[4];
  const float* V1    = (const float*)d_in[5];
  const float* coef1 = (const float*)d_in[6];
  const float* bias1 = (const float*)d_in[7];
  const float* V2    = (const float*)d_in[8];
  const float* coef2 = (const float*)d_in[9];
  const float* bias2 = (const float*)d_in[10];
  float* out = (float*)d_out;

  char* ws = (char*)d_ws;
  size_t off = 0;
  auto walloc = [&](size_t bytes) -> void* {
    void* p = ws + off;
    off += (bytes + 255) & ~(size_t)255;
    return p;
  };
  float*    h1acc = (float*)   walloc(sizeof(float) * N_NODES * HD);      // 25.6 MB
  _Float16* Wt1   = (_Float16*)walloc(sizeof(_Float16) * N_RELS * HD * HD);
  _Float16* Wt2   = (_Float16*)walloc(sizeof(_Float16) * N_RELS * HD * HD);
  int* order  = (int*)walloc(sizeof(int) * N_EDGES);
  int* counts = (int*)walloc(sizeof(int) * N_RELS);
  int* roff   = (int*)walloc(sizeof(int) * N_RELS);
  int* cursor = (int*)walloc(sizeof(int) * N_RELS);

  const int NH = N_NODES * HD;

  init_zero_kernel<<<(2 * NH + 255) / 256, 256, 0, stream>>>(h1acc, out, counts, NH);

  build_w_kernel<<<(N_RELS * HD * HD + 255) / 256, 256, 0, stream>>>(V1, coef1, Wt1);
  build_w_kernel<<<(N_RELS * HD * HD + 255) / 256, 256, 0, stream>>>(V2, coef2, Wt2);

  hist_kernel<<<(N_EDGES + 255) / 256, 256, 0, stream>>>(etype, counts);
  prefix_kernel<<<1, 1, 0, stream>>>(counts, roff, cursor);
  scatter_kernel<<<(N_EDGES + 255) / 256, 256, 0, stream>>>(etype, cursor, order);

  dim3 gg((N_EDGES + 63) / 64, N_RELS);

  // layer 1: h1 = relu(segsum(norm * h[src] @ W1[etype]) + bias1)
  rgcn_gemm_kernel<<<gg, 256, 0, stream>>>(h, Wt1, order, roff, counts, src, dst, norm, h1acc);
  bias_relu_kernel<<<(NH + 255) / 256, 256, 0, stream>>>(h1acc, bias1, NH);

  // layer 2: out = segsum(norm * h1[src] @ W2[etype]) + bias2
  rgcn_gemm_kernel<<<gg, 256, 0, stream>>>(h1acc, Wt2, order, roff, counts, src, dst, norm, out);
  bias_kernel<<<(NH + 255) / 256, 256, 0, stream>>>(out, bias2, NH);

  (void)in_sizes; (void)n_in; (void)out_size; (void)ws_size;
}
